// VQAttention_2568390443469
// MI455X (gfx1250) — matrix operand order repaired and checked by reading, not compiled, not run on hardware
//
#include <hip/hip_runtime.h>
#include <hip/hip_bf16.h>
#include <stdint.h>

// ---------------- problem constants ----------------
#define BB   2048
#define LC   20
#define LQ   50
#define CTX  2048
#define CTXM 512
#define NEGV (-1.0e30f)

// LDS row strides (bank-conflict avoidance on 64 banks)
#define LDV 516
#define LDQ 516
#define LDS1 53

typedef __attribute__((ext_vector_type(16))) __bf16 v16bf;
typedef __attribute__((ext_vector_type(8)))  float  v8f;

union FragU  { uint32_t u[8]; v16bf v; };
union FragU4 { uint4 q[2]; uint32_t u[8]; v16bf v; };

__device__ __forceinline__ uint32_t bf16_rne(float x) {
  uint32_t u = __float_as_uint(x);
  return (u + 0x7FFFu + ((u >> 16) & 1u)) >> 16;
}
__device__ __forceinline__ uint32_t pack_bf16(float lo, float hi) {
  return bf16_rne(lo) | (bf16_rne(hi) << 16);
}

// ---------------------------------------------------------------
// Kernel 1: W = trans_g * trans_v / ||trans_v||_row, emitted as bf16
// in the B-fragment register order this code uses (contiguous k, same as A):
//   wtile[kchunk][ntile][lane][v]  (dwords), kchunk=K/32, ntile=N/16
// lane L: n = ntile*16 + (L&15); dword v holds {W[n][k], W[n][k+1]}
// with k = kchunk*32 + (L>=16 ? 16 : 0) + 2*v.
// => each lane's fragment is 32 contiguous bytes -> 2x global_load_b128.
// ---------------------------------------------------------------
__global__ __launch_bounds__(256)
void vq_prep_w(const float* __restrict__ trans_v,
               const float* __restrict__ trans_g,
               uint32_t* __restrict__ wpk) {
  const int m = blockIdx.x;           // output row of W (the WMMA "n")
  const int t = threadIdx.x;
  __shared__ float red[256];
  const float* row = trans_v + (size_t)m * CTX;
  float ss = 0.f;
  for (int c = t; c < CTX; c += 256) { float x = row[c]; ss += x * x; }
  red[t] = ss;
  __syncthreads();
  for (int s = 128; s > 0; s >>= 1) {
    if (t < s) red[t] += red[t + s];
    __syncthreads();
  }
  const float scale = trans_g[m] / sqrtf(red[0]);

  const int ntile = m >> 4;
  const int ln    = m & 15;
  // 64 kchunks x 2 lane-halves x 8 dwords = 1024 dwords for this m
  for (int i = t; i < 1024; i += 256) {
    int kchunk = i >> 4;
    int rem    = i & 15;
    int halfs  = rem >> 3;
    int v      = rem & 7;
    int lane   = ln + halfs * 16;
    int k      = kchunk * 32 + halfs * 16 + 2 * v;
    uint32_t pk = pack_bf16(row[k] * scale, row[k + 1] * scale);
    wpk[(size_t)(((kchunk * 32 + ntile) * 32) + lane) * 8 + v] = pk;
  }
}

// ---------------------------------------------------------------
// Kernel 2: fully fused per-batch pipeline. 1 block = 1 batch,
// 256 threads = 8 wave32s. ~165 KB LDS -> 1 WG per WGP.
// ---------------------------------------------------------------
__global__ __launch_bounds__(256)
void vq_fused(const float* __restrict__ vid,      // (B, LC, CTX)
              const float* __restrict__ ques,     // (B, CTXM, LQ)
              const float* __restrict__ vmask,    // (B, LC)
              const float* __restrict__ qmask,    // (B, LQ)
              const float* __restrict__ trans_b,  // (CTXM)
              const float* __restrict__ w4V,      // (CTXM)
              const float* __restrict__ w4Q,      // (CTXM)
              const float* __restrict__ w4mlu,    // (CTXM)
              const float* __restrict__ bias,     // (1)
              const uint32_t* __restrict__ wpk,   // tiled bf16 W (see prep)
              float* __restrict__ out)            // (B, 4*CTXM, LC)
{
  __shared__ uint32_t s_apk[2][32 * 17];   // double-buffered stage-1 A chunk
  __shared__ float s_V[LC * LDV];          // V = Vid@W^T + b   (f32)
  __shared__ float s_Q[LQ * LDQ];          // Q[q][d]           (f32)
  __shared__ float s_w4V[CTXM], s_w4Q[CTXM], s_w4mlu[CTXM];
  __shared__ float s_s0[LC], s_s1[LQ];
  __shared__ float s_S[LC * LDS1], s_S1[LC * LDS1], s_S2[LC * LDS1];
  __shared__ float s_Bt[LC * LC];

  const int b    = blockIdx.x;
  const int t    = threadIdx.x;
  const int lane = t & 31;
  const int wid  = t >> 5;
  const int half = lane >> 4;   // 0: lanes 0-15, 1: lanes 16-31
  const int ln   = lane & 15;

  const size_t vbase = (size_t)b * LC * CTX;
  const size_t qbase = (size_t)b * CTXM * LQ;

  // ---- Stage 0: stage small vectors + Q (coalesced global read) ----
  for (int i = t; i < CTXM; i += 256) {
    s_w4V[i]   = w4V[i];
    s_w4Q[i]   = w4Q[i];
    s_w4mlu[i] = w4mlu[i];
  }
  for (int i = t; i < CTXM * LQ; i += 256) {
    int d = i / LQ, q = i % LQ;
    s_Q[q * LDQ + d] = ques[qbase + i];
  }

  // ---- Stage 1: V = Vid @ W^T + trans_b via bf16 WMMA ----
  // 2 M-tiles (rows 0-15 / 16-31, rows>=20 zero) x 32 N-tiles.
  // wave w owns N-tiles [4w, 4w+4); both M-tiles share B fragments.
  const v8f vzero = {0.f,0.f,0.f,0.f,0.f,0.f,0.f,0.f};
  v8f acc0[4], acc1[4];
  for (int nt = 0; nt < 4; ++nt) { acc0[nt] = vzero; acc1[nt] = vzero; }

  // chunk filler: 32 rows x 16 k-pairs of Vid as packed bf16
  auto fill_chunk = [&](int kb, int buf) {
    for (int i = t; i < 512; i += 256) {
      int r = i >> 4, p = i & 15;
      uint32_t pk = 0;
      if (r < LC) {
        const float* src = vid + vbase + (size_t)r * CTX + kb + 2 * p;
        pk = pack_bf16(src[0], src[1]);
      }
      s_apk[buf][r * 17 + p] = pk;
    }
  };

  fill_chunk(0, 0);
  const int NKI = CTX / 32;   // 64
  for (int it = 0; it < NKI; ++it) {
    __syncthreads();                     // chunk `it` staged; prev reads done
    if (it + 1 < NKI) fill_chunk((it + 1) * 32, (it + 1) & 1);  // prefetch

    const int buf = it & 1;
    const int kchunk = it;

    // Load ALL four B fragments first (distinct regs -> staggered waits,
    // one 8-wide global_load_b128 clause overlapping the WMMAs).
    FragU4 bu[4];
#pragma unroll
    for (int nt = 0; nt < 4; ++nt) {
      const int ntile = wid * 4 + nt;
      const uint4* bp = (const uint4*)
          (wpk + (size_t)(((kchunk * 32 + ntile) * 32) + lane) * 8);
      bu[nt].q[0] = bp[0];
      bu[nt].q[1] = bp[1];
    }

    // A fragments (contiguous k = 16*half + s, matching packed B)
    v16bf afrag[2];
#pragma unroll
    for (int mt = 0; mt < 2; ++mt) {
      const int r = mt * 16 + ln;
      FragU au;
#pragma unroll
      for (int v = 0; v < 8; ++v) {
        int pair = v + (half << 3);
        au.u[v] = s_apk[buf][r * 17 + pair];
      }
      afrag[mt] = au.v;
    }

#pragma unroll
    for (int nt = 0; nt < 4; ++nt) {
      acc0[nt] = __builtin_amdgcn_wmma_f32_16x16x32_bf16(
          false, afrag[0], false, bu[nt].v, (short)0, acc0[nt], false, false);
      acc1[nt] = __builtin_amdgcn_wmma_f32_16x16x32_bf16(
          false, afrag[1], false, bu[nt].v, (short)0, acc1[nt], false, false);
    }
  }
  __syncthreads();

  // C/D layout: VGPR j -> row j (lanes 0-15) / row j+8 (lanes 16-31)
#pragma unroll
  for (int nt = 0; nt < 4; ++nt) {
    const int col = wid * 64 + nt * 16 + ln;
    const float tb = trans_b[col];
#pragma unroll
    for (int j = 0; j < 8; ++j) {
      int r0 = j + half * 8;           // mt = 0
      if (r0 < LC) s_V[r0 * LDV + col] = acc0[nt][j] + tb;
      int r1 = 16 + j + half * 8;      // mt = 1
      if (r1 < LC) s_V[r1 * LDV + col] = acc1[nt][j] + tb;
    }
  }
  __syncthreads();

  // ---- Stage 2a: s0[l] = V[l]·w4V ; s1[q] = Q[q]·w4Q ----
  for (int l = wid; l < LC; l += 8) {
    float s = 0.f;
    for (int d = lane; d < CTXM; d += 32) s += s_V[l * LDV + d] * s_w4V[d];
#pragma unroll
    for (int off = 16; off > 0; off >>= 1) s += __shfl_xor(s, off, 32);
    if (lane == 0) s_s0[l] = s;
  }
  for (int q = wid; q < LQ; q += 8) {
    float s = 0.f;
    for (int d = lane; d < CTXM; d += 32) s += s_Q[q * LDQ + d] * s_w4Q[d];
#pragma unroll
    for (int off = 16; off > 0; off >>= 1) s += __shfl_xor(s, off, 32);
    if (lane == 0) s_s1[q] = s;
  }
  __syncthreads();

  // ---- Stage 2b: s2 = (V*w4mlu) @ Q^T via WMMA; wave w -> tile (mt,nt) ----
  {
    const int mt = wid & 1, nt = wid >> 1;   // 2 M-tiles x 4 N-tiles (q pad 64)
    const int l0 = mt * 16, q0 = nt * 16;
    v8f sacc = vzero;
    const int ra = l0 + ln;
    const int qb = q0 + ln;
    for (int kb = 0; kb < CTXM; kb += 32) {
      FragU au, bu;
#pragma unroll
      for (int v = 0; v < 8; ++v) {
        int pair = v + (half << 3);
        int d = kb + 2 * pair;
        float f0 = (ra < LC) ? s_V[ra * LDV + d]     * s_w4mlu[d]     : 0.f;
        float f1 = (ra < LC) ? s_V[ra * LDV + d + 1] * s_w4mlu[d + 1] : 0.f;
        au.u[v] = pack_bf16(f0, f1);
        int pb = v + (half << 3);
        int db = kb + 2 * pb;
        float g0 = (qb < LQ) ? s_Q[qb * LDQ + db]     : 0.f;
        float g1 = (qb < LQ) ? s_Q[qb * LDQ + db + 1] : 0.f;
        bu.u[v] = pack_bf16(g0, g1);
      }
      sacc = __builtin_amdgcn_wmma_f32_16x16x32_bf16(
          false, au.v, false, bu.v, (short)0, sacc, false, false);
    }
    const float bv = bias[0];
#pragma unroll
    for (int j = 0; j < 8; ++j) {
      int l = l0 + j + half * 8;
      int q = q0 + ln;
      if (l < LC && q < LQ)
        s_S[l * LDS1 + q] = sacc[j] + s_s0[l] + s_s1[q] + bv;
    }
  }
  __syncthreads();

  // ---- Stage 3: dual masked softmax ----
  // S1: softmax over q (one wave per row, 2 q per lane)
  for (int l = wid; l < LC; l += 8) {
    int q1 = lane, q2 = lane + 32;
    float qm1 = qmask[(size_t)b * LQ + q1];
    float x1 = s_S[l * LDS1 + q1] * qm1 + (1.f - qm1) * NEGV;
    float x2 = -3.4e38f;
    if (q2 < LQ) {
      float qm2 = qmask[(size_t)b * LQ + q2];
      x2 = s_S[l * LDS1 + q2] * qm2 + (1.f - qm2) * NEGV;
    }
    float mx = fmaxf(x1, x2);
#pragma unroll
    for (int off = 16; off > 0; off >>= 1) mx = fmaxf(mx, __shfl_xor(mx, off, 32));
    float e1 = expf(x1 - mx);
    float e2 = (q2 < LQ) ? expf(x2 - mx) : 0.f;
    float sm = e1 + e2;
#pragma unroll
    for (int off = 16; off > 0; off >>= 1) sm += __shfl_xor(sm, off, 32);
    float inv = 1.f / sm;
    s_S1[l * LDS1 + q1] = e1 * inv;
    if (q2 < LQ) s_S1[l * LDS1 + q2] = e2 * inv;
  }
  // S2: softmax over l (one thread per column q)
  if (t < LQ) {
    const int q = t;
    float mx = -3.4e38f;
    for (int l = 0; l < LC; ++l) {
      float vm = vmask[(size_t)b * LC + l];
      float x = s_S[l * LDS1 + q] * vm + (1.f - vm) * NEGV;
      mx = fmaxf(mx, x);
    }
    float sm = 0.f;
    for (int l = 0; l < LC; ++l) {
      float vm = vmask[(size_t)b * LC + l];
      float x = s_S[l * LDS1 + q] * vm + (1.f - vm) * NEGV;
      sm += expf(x - mx);
    }
    float inv = 1.f / sm;
    for (int l = 0; l < LC; ++l) {
      float vm = vmask[(size_t)b * LC + l];
      float x = s_S[l * LDS1 + q] * vm + (1.f - vm) * NEGV;
      s_S2[l * LDS1 + q] = expf(x - mx) * inv;
    }
  }
  __syncthreads();

  // ---- Stage 4: Bt[l][m] = sum_q S1[l][q] * S2[m][q]  (20x20) ----
  for (int i = t; i < LC * LC; i += 256) {
    int l = i / LC, m = i % LC;
    float s = 0.f;
    for (int q = 0; q < LQ; ++q) s += s_S1[l * LDS1 + q] * s_S2[m * LDS1 + q];
    s_Bt[i] = s;
  }
  __syncthreads();

  // ---- Stage 5: A, Bv, and transposed concat output (B, 2048, LC) ----
  const size_t ob = (size_t)b * (4 * CTXM) * LC;
  for (int d0 = 0; d0 < CTXM; d0 += 8) {
    const int l = t & 31;
    const int d = d0 + (t >> 5);
    if (l < LC) {
      float a = 0.f;
      for (int q = 0; q < LQ; ++q) a += s_S1[l * LDS1 + q] * s_Q[q * LDQ + d];
      float bv = 0.f;
      for (int m = 0; m < LC; ++m) bv += s_Bt[l * LC + m] * s_V[m * LDV + d];
      const float v = s_V[l * LDV + d];
      out[ob + (size_t)(d)          * LC + l] = v;
      out[ob + (size_t)(CTXM + d)   * LC + l] = a;
      out[ob + (size_t)(2*CTXM + d) * LC + l] = v * a;
      out[ob + (size_t)(3*CTXM + d) * LC + l] = v * bv;
    }
  }
}

// ---------------------------------------------------------------
extern "C" void kernel_launch(void* const* d_in, const int* in_sizes, int n_in,
                              void* d_out, int out_size, void* d_ws, size_t ws_size,
                              hipStream_t stream) {
  const float* vid     = (const float*)d_in[0];
  const float* ques    = (const float*)d_in[1];
  const float* vmask   = (const float*)d_in[2];
  const float* qmask   = (const float*)d_in[3];
  const float* trans_v = (const float*)d_in[4];
  const float* trans_g = (const float*)d_in[5];
  const float* trans_b = (const float*)d_in[6];
  const float* w4V     = (const float*)d_in[7];
  const float* w4Q     = (const float*)d_in[8];
  const float* w4mlu   = (const float*)d_in[9];
  const float* bias    = (const float*)d_in[10];
  float* out = (float*)d_out;

  uint32_t* wpk = (uint32_t*)d_ws;  // 64*32*32*8 dwords = 2 MB

  vq_prep_w<<<CTXM, 256, 0, stream>>>(trans_v, trans_g, wpk);
  vq_fused<<<BB, 256, 0, stream>>>(vid, ques, vmask, qmask, trans_b,
                                   w4V, w4Q, w4mlu, bias, wpk, out);
}
